// P20SequencePrimitiveBranch_47064251630316
// MI455X (gfx1250) — matrix operand order repaired and checked by reading, not compiled, not run on hardware
//
#include <hip/hip_runtime.h>

// ---------------- problem constants ----------------
constexpr int B_  = 4;
constexpr int S_  = 4096;
constexpr int D_  = 1024;
constexpr int DW_ = 512;
constexpr int NB_ = 256;           // DW_/2 rotation blocks
constexpr int CHUNKS = 64;         // parallel-scan chunks
constexpr int CLEN   = S_ / CHUNKS;// 64 steps per chunk
constexpr int M_  = B_ * S_;       // 16384 GEMM rows
constexpr int KT  = 64;            // K chunk staged in LDS per block

// ---------------- WMMA vector types ----------------
typedef __attribute__((ext_vector_type(16))) __bf16 v16bf;
typedef __attribute__((ext_vector_type(8)))  float  v8f;

// float -> bf16 round-to-nearest-even
__device__ __forceinline__ unsigned short f2bf(float f) {
    union { float f; unsigned int u; } x; x.f = f;
    unsigned int u = x.u + 0x7FFFu + ((x.u >> 16) & 1u);
    return (unsigned short)(u >> 16);
}

// ---------------- f32 -> bf16 bulk conversion (vectorized) ----------------
__global__ __launch_bounds__(256)
void cvt_f32_bf16_v4(const float4* __restrict__ src,
                     ushort4* __restrict__ dst, int n4) {
    int i = blockIdx.x * blockDim.x + threadIdx.x;
    if (i < n4) {
        float4 v = src[i];
        ushort4 o;
        o.x = f2bf(v.x); o.y = f2bf(v.y); o.z = f2bf(v.z); o.w = f2bf(v.w);
        dst[i] = o;
    }
}

// ---------------- GEMM:  C[M,N] = A[M,K] * W[N,K]^T  (all-bf16 inputs) ----
// Block = 8 waves sharing one 32-row A strip staged through LDS
// (double-buffered 64-wide K chunks). Each wave owns a 32x64 output tile:
// 2 A frags (ds_load_b128) x 4 B frags (global_load_b128)
// -> 8 v_wmma_f32_16x16x32_bf16 per 32-wide K step.
//
// Fragment layouts per CDNA5 ISA 7.12.2:
//  A 16x32 bf16 : lane (m = l&15, h = l>>4) holds K runs [k0+8h, +7] and
//                 [k0+8h+16, +7]
//  B 32x16 bf16 : lane n = l&15 holds K runs [k0+8h, +7] and [k0+8h+16, +7]
//  C 16x16 f32  : VGPR r -> row r + 8h, col = l&15
__device__ __forceinline__ v16bf load_frag_b(const unsigned short* p) {
    union { v16bf v; int4 q[2]; } r;
    r.q[0] = *(const int4*)(p);        // K 0..7
    r.q[1] = *(const int4*)(p + 16);   // K 16..23
    return r.v;
}

__global__ __launch_bounds__(256)
void gemm_bf16_wmma(const unsigned short* __restrict__ A,   // [M,K] bf16
                    const unsigned short* __restrict__ W,   // [N,K] bf16
                    float* __restrict__ C,                  // [M,N] f32
                    int M, int N, int K) {
    __shared__ unsigned short As[2][32 * KT];   // 8 KB double buffer

    const int wid  = threadIdx.x >> 5;          // wave in block: 0..7
    const int lane = threadIdx.x & 31;
    const int tiles_n = N >> 6;                 // 64-wide column tiles
    const int bpr  = tiles_n >> 3;              // blocks per tile-row
    const int tm   = blockIdx.x / bpr;
    const int tn   = (blockIdx.x % bpr) * 8 + wid;
    const int row0 = tm << 5;                   // 32 shared rows per block
    const int col0 = tn << 6;
    const int m    = lane & 15;
    const int half = lane >> 4;

    // cooperative A stager: 256 threads x 16B == 32 rows x 64 bf16
    const int srow = threadIdx.x >> 3;          // 0..31
    const int sseg = (threadIdx.x & 7) * 8;     // 0,8,..,56 (bf16 elems)
    const unsigned short* sbase = A + (size_t)(row0 + srow) * K + sseg;

    auto stage = [&](int buf, int kc) {
        __builtin_prefetch((const void*)(sbase + kc + KT), 0, 3); // 2 chunks ahead
        *(int4*)&As[buf][srow * KT + sseg] = *(const int4*)(sbase + kc);
    };

    v8f acc0[4] = { v8f{}, v8f{}, v8f{}, v8f{} };   // rows row0..row0+15
    v8f acc1[4] = { v8f{}, v8f{}, v8f{}, v8f{} };   // rows row0+16..+31

    stage(0, 0);
    __syncthreads();

    for (int kc = 0; kc < K; kc += KT) {
        const int buf = (kc / KT) & 1;
        if (kc + KT < K) stage(buf ^ 1, kc + KT);

#pragma unroll
        for (int kk = 0; kk < KT; kk += 32) {
            // A fragments from LDS
            union { v16bf v; int4 q[2]; } a0, a1;
            const unsigned short* pa0 = &As[buf][(m)      * KT + kk + half * 8];
            const unsigned short* pa1 = &As[buf][(16 + m) * KT + kk + half * 8];
            a0.q[0] = *(const int4*)(pa0);
            a0.q[1] = *(const int4*)(pa0 + 16);
            a1.q[0] = *(const int4*)(pa1);
            a1.q[1] = *(const int4*)(pa1 + 16);

#pragma unroll
            for (int t = 0; t < 4; ++t) {
                const unsigned short* pb =
                    W + (size_t)(col0 + t * 16 + m) * K + kc + kk + half * 8;
                v16bf bf = load_frag_b(pb);
                acc0[t] = __builtin_amdgcn_wmma_f32_16x16x32_bf16(
                    false, a0.v, false, bf, (short)0, acc0[t], false, false);
                acc1[t] = __builtin_amdgcn_wmma_f32_16x16x32_bf16(
                    false, a1.v, false, bf, (short)0, acc1[t], false, false);
            }
        }
        __syncthreads();   // reads of buf done; stage of buf^1 visible
    }

#pragma unroll
    for (int t = 0; t < 4; ++t) {
#pragma unroll
        for (int r = 0; r < 8; ++r) {
            C[(size_t)(row0 + half * 8 + r) * N + col0 + t * 16 + m]      = acc0[t][r];
            C[(size_t)(row0 + 16 + half * 8 + r) * N + col0 + t * 16 + m] = acc1[t][r];
        }
    }
}

// ---------------- chunked linear scan (rotation SSM), fp32 ----------------
// Phase 1: per-chunk local scan with zero init; emit chunk-end carry.
__global__ __launch_bounds__(256)
void scan_local(float* __restrict__ xbuf,          // [B,S,DW] in/out
                float* __restrict__ carry,         // [B,CHUNKS,NB] float2
                const float* __restrict__ theta) {
    int tid = blockIdx.x * blockDim.x + threadIdx.x;   // B*NB*CHUNKS threads
    int b     = tid >> 14;            // / (NB_*CHUNKS)
    int rem   = tid & 16383;
    int chunk = rem >> 8;             // / NB_
    int nb    = rem & 255;

    float th = theta[nb];
    float ct = __cosf(th), st = __sinf(th);

    size_t base = ((size_t)b * S_ + (size_t)chunk * CLEN) * DW_ + 2 * nb;
    float sx = 0.f, sy = 0.f;
#pragma unroll 4
    for (int t = 0; t < CLEN; ++t) {
        float2 xt = *(const float2*)(xbuf + base + (size_t)t * DW_);
        float nx = ct * sx - st * sy + xt.x;
        float ny = st * sx + ct * sy + xt.y;
        sx = nx; sy = ny;
        *(float2*)(xbuf + base + (size_t)t * DW_) = make_float2(sx, sy);
    }
    ((float2*)carry)[((size_t)b * CHUNKS + chunk) * NB_ + nb] = make_float2(sx, sy);
}

// Phase 2: sequential combine of chunk carries; overwrite with exclusive
// prefix (carry-IN per chunk). R^CLEN is rotation by CLEN*theta (exact).
__global__ __launch_bounds__(256)
void scan_carry(float* __restrict__ carry, const float* __restrict__ theta) {
    int tid = blockIdx.x * blockDim.x + threadIdx.x;   // B*NB threads
    int b  = tid >> 8;
    int nb = tid & 255;
    float th = theta[nb];
    float cL = __cosf((float)CLEN * th), sL = __sinf((float)CLEN * th);

    float2* cr = (float2*)carry;
    float cx = 0.f, cy = 0.f;
    for (int c = 0; c < CHUNKS; ++c) {
        size_t idx = ((size_t)b * CHUNKS + c) * NB_ + nb;
        float2 cout = cr[idx];
        cr[idx] = make_float2(cx, cy);     // exclusive prefix = carry-in
        float nx = cL * cx - sL * cy + cout.x;
        float ny = sL * cx + cL * cy + cout.y;
        cx = nx; cy = ny;
    }
}

// Phase 3: add R^{t+1} * carry_in to each local result and EMIT bf16 mixed
// (fused conversion for the up-projection GEMM; f32 buffer is not rewritten).
__global__ __launch_bounds__(256)
void scan_fix(const float* __restrict__ xbuf,
              const float* __restrict__ carry,
              const float* __restrict__ theta,
              unsigned int* __restrict__ mixed_bf) {   // [B,S,DW/2] packed 2xbf16
    int tid = blockIdx.x * blockDim.x + threadIdx.x;   // B*NB*CHUNKS threads
    int b     = tid >> 14;
    int rem   = tid & 16383;
    int chunk = rem >> 8;
    int nb    = rem & 255;

    float th = theta[nb];
    float ct = __cosf(th), st = __sinf(th);

    float2 cin = ((const float2*)carry)[((size_t)b * CHUNKS + chunk) * NB_ + nb];
    // v = R^1 * carry_in
    float vx = ct * cin.x - st * cin.y;
    float vy = st * cin.x + ct * cin.y;

    size_t base  = ((size_t)b * S_ + (size_t)chunk * CLEN) * DW_ + 2 * nb;
    size_t baseh = ((size_t)b * S_ + (size_t)chunk * CLEN) * (DW_ / 2) + nb;
#pragma unroll 4
    for (int t = 0; t < CLEN; ++t) {
        float2 mv = *(const float2*)(xbuf + base + (size_t)t * DW_);
        float ox = mv.x + vx;
        float oy = mv.y + vy;
        mixed_bf[baseh + (size_t)t * (DW_ / 2)] =
            (unsigned int)f2bf(ox) | ((unsigned int)f2bf(oy) << 16);
        float nx = ct * vx - st * vy;
        float ny = st * vx + ct * vy;
        vx = nx; vy = ny;
    }
}

// ---------------- launch ----------------
extern "C" void kernel_launch(void* const* d_in, const int* in_sizes, int n_in,
                              void* d_out, int out_size, void* d_ws, size_t ws_size,
                              hipStream_t stream) {
    (void)in_sizes; (void)n_in; (void)out_size; (void)ws_size;

    const float* hidden = (const float*)d_in[0];   // [B,S,D]
    const float* W_in   = (const float*)d_in[1];   // [DW,D]
    const float* theta  = (const float*)d_in[2];   // [NB]
    const float* W_out  = (const float*)d_in[3];   // [D,DW]
    float* out = (float*)d_out;                    // [B,S,D]

    // workspace layout (all offsets multiples of 16B)
    float* xbuf  = (float*)d_ws;                                   // 32 MB f32 x/local-scan
    float* carry = xbuf + (size_t)B_ * S_ * DW_;                   // 512 KB float2 carries
    unsigned short* wb_in  = (unsigned short*)(carry + (size_t)B_ * CHUNKS * NB_ * 2);
    unsigned short* wb_out = wb_in  + (size_t)DW_ * D_;            // 1 MB each (bf16 weights)
    unsigned short* hid_bf = wb_out + (size_t)D_ * DW_;            // 32 MB bf16 hidden
    unsigned short* mix_bf = hid_bf + (size_t)M_ * D_;             // 16 MB bf16 mixed

    // 0) bulk f32 -> bf16 conversions (weights + hidden)
    {
        int nw4 = (DW_ * D_) / 4;                  // 131072
        cvt_f32_bf16_v4<<<(nw4 + 255) / 256, 256, 0, stream>>>(
            (const float4*)W_in,  (ushort4*)wb_in,  nw4);
        cvt_f32_bf16_v4<<<(nw4 + 255) / 256, 256, 0, stream>>>(
            (const float4*)W_out, (ushort4*)wb_out, nw4);
        int nh4 = (M_ * D_) / 4;                   // 4194304
        cvt_f32_bf16_v4<<<(nh4 + 255) / 256, 256, 0, stream>>>(
            (const float4*)hidden, (ushort4*)hid_bf, nh4);
    }

    // 1) down-projection: xbuf[M,DW] = hidden[M,D] * W_in[DW,D]^T
    {
        int blocks = (M_ / 32) * ((DW_ / 64) / 8);   // 512
        gemm_bf16_wmma<<<blocks, 256, 0, stream>>>(hid_bf, wb_in, xbuf,
                                                   M_, DW_, D_);
    }

    // 2) chunked parallel scan over S (fp32, in place on xbuf)
    scan_local<<<(B_ * NB_ * CHUNKS) / 256, 256, 0, stream>>>(xbuf, carry, theta);
    scan_carry<<<(B_ * NB_) / 256, 256, 0, stream>>>(carry, theta);
    scan_fix  <<<(B_ * NB_ * CHUNKS) / 256, 256, 0, stream>>>(
        xbuf, carry, theta, (unsigned int*)mix_bf);

    // 3) up-projection: out[M,D] = mixed[M,DW] * W_out[D,DW]^T
    {
        int blocks = (M_ / 32) * ((D_ / 64) / 8);    // 1024
        gemm_bf16_wmma<<<blocks, 256, 0, stream>>>(mix_bf, wb_out, out,
                                                   M_, D_, DW_);
    }
}
